// TransformerEncoderLayer_2869038154481
// MI455X (gfx1250) — matrix operand order repaired and checked by reading, not compiled, not run on hardware
//
#include <hip/hip_runtime.h>
#include <hip/hip_bf16.h>
#include <stdint.h>
#include <stddef.h>

// Problem sizes (compile-time, from reference)
#define TB  4
#define TS  2048
#define TD  1024
#define THD 16
#define TF  4096
#define TDH 64

typedef __bf16 bf16;
typedef __attribute__((ext_vector_type(16))) __bf16 v16bf;
typedef __attribute__((ext_vector_type(8)))  __bf16 v8bf;
typedef __attribute__((ext_vector_type(8)))  float  v8f;

typedef __attribute__((ext_vector_type(4))) unsigned int u32x4;
typedef __attribute__((ext_vector_type(4))) int          i32x4;
typedef __attribute__((ext_vector_type(8))) int          i32x8;

#if defined(__has_builtin)
#if __has_builtin(__builtin_amdgcn_tensor_load_to_lds) && \
    __has_builtin(__builtin_amdgcn_s_wait_tensorcnt)
#define HAVE_TDM 1
#endif
#endif
#ifndef HAVE_TDM
#define HAVE_TDM 0
#endif

union Frag16 { v16bf v; v8bf h[2]; };

__device__ __forceinline__ bf16 f2bf(float f) {
  uint32_t u = __builtin_bit_cast(uint32_t, f);
  u += 0x7fffu + ((u >> 16) & 1u);          // round-to-nearest-even
  return __builtin_bit_cast(bf16, (uint16_t)(u >> 16));
}

__device__ __forceinline__ v8f wmma_bf16(const Frag16& a, const Frag16& b, v8f c) {
  // D(16x16 f32) = A(16x32 bf16) x B(32x16 bf16) + C
  return __builtin_amdgcn_wmma_f32_16x16x32_bf16(false, a.v, false, b.v,
                                                 (short)0, c, false, false);
}

#if HAVE_TDM
// Issue a TDM 2D tile load: global (row-major, elem=2B) -> LDS, with LDS row
// padding of 4 DWORDs every 16 DWORDs (i.e. 32 bf16 data + 8 bf16 pad = our
// 40-element LDS row stride). tile_dim0 = 32 elems (one K-row), tile_dim1 =
// number of rows. Wave-level op, tracked by TENSORcnt.
__device__ __forceinline__ void tdm_load_tile_2d(uint32_t lds_byte_addr,
                                                 const void* gaddr,
                                                 uint32_t tensor_d0,
                                                 uint32_t tensor_d1,
                                                 uint64_t row_stride_elems,
                                                 uint32_t tile_d0,
                                                 uint32_t tile_d1) {
  const uint64_t ga = (uint64_t)(uintptr_t)gaddr;
  u32x4 g0;
  g0[0] = 1u;                                   // count=1, user descriptor
  g0[1] = lds_byte_addr;                        // LDS destination (bytes)
  g0[2] = (uint32_t)(ga & 0xffffffffu);         // global_addr[31:0]
  g0[3] = (uint32_t)((ga >> 32) & 0x01ffffffu)  // global_addr[56:32]
          | (2u << 30);                         // type = 2 ("image")

  const uint64_t s0 = row_stride_elems;         // tensor_dim0_stride (elems)
  i32x8 g1;
  g1[0] = (int)((1u << 16)                      // data_size = 1 -> 2 bytes
              | (1u << 20)                      // pad_enable
              | (3u << 22)                      // pad_interval: 16 DWORDs
              | (3u << 25));                    // pad_amount: 4 DWORDs
  g1[1] = (int)((tensor_d0 & 0xffffu) << 16);   // [15:0] abar=0, [31:16] d0.lo
  g1[2] = (int)((tensor_d0 >> 16) | ((tensor_d1 & 0xffffu) << 16));
  g1[3] = (int)((tensor_d1 >> 16) | (tile_d0 << 16));   // tile_dim0
  g1[4] = (int)(tile_d1 & 0xffffu);             // tile_dim1 (tile_dim2 = 0)
  g1[5] = (int)(uint32_t)(s0 & 0xffffffffu);    // dim0_stride[31:0]
  g1[6] = (int)(uint32_t)((s0 >> 32) & 0xffffu);// dim0_stride[47:32], d1s=0
  g1[7] = 0;
  const i32x4 z4 = {0, 0, 0, 0};
#if __clang_major__ >= 23
  const i32x8 z8 = {0, 0, 0, 0, 0, 0, 0, 0};
  __builtin_amdgcn_tensor_load_to_lds(g0, g1, z4, z4, z8, 0);
#else
  __builtin_amdgcn_tensor_load_to_lds(g0, g1, z4, z4, 0);
#endif
}
#endif  // HAVE_TDM

// ---------------------------------------------------------------- fp32->bf16
__global__ void __launch_bounds__(256)
cvt_f2bf4(const float* __restrict__ in, bf16* __restrict__ out, long n) {
  long i = ((long)blockIdx.x * 256 + threadIdx.x) * 4;
  if (i + 3 >= n + 4) return;  // sizes are multiples of 1024; guard anyway
  float4 f = *(const float4*)(in + i);
  out[i + 0] = f2bf(f.x);
  out[i + 1] = f2bf(f.y);
  out[i + 2] = f2bf(f.z);
  out[i + 3] = f2bf(f.w);
}

// ---------------------------------------------------------------- GEMM
// C[M,N] = A[M,K] @ B[K,N] + bias  (A,B bf16 row-major; fp32 accumulate)
// Block: 256 threads (8 waves). Tile: BM=128, BN=128, BK=32.
// A tile staged by the Tensor Data Mover (wave 0 issues, TENSORcnt-tracked);
// B tile staged transposed by all threads. Wave (wm,wn) owns 2x4 WMMA tiles.
template <bool RELU, bool OUTBF>
__global__ void __launch_bounds__(256)
gemm_bf16k(const bf16* __restrict__ A, const bf16* __restrict__ Bm,
           const float* __restrict__ bias, void* __restrict__ Cout,
           int M, int N, int K) {
  constexpr int BM = 128, BN = 128, BK = 32, LDT = BK + 8;  // pad: stride 40
  __shared__ __align__(16) bf16 As[BM][LDT];
  __shared__ __align__(16) bf16 Bs[BN][LDT];  // transposed: [col][k]

  const int tid  = threadIdx.x;
  const int wid  = tid >> 5;
  const int lane = tid & 31;
  const int ln   = lane & 15;
  const int hi   = lane >> 4;
  const int o    = hi << 3;             // A-frag K offset {0,8}
  const int m_base = blockIdx.y * BM;
  const int n_base = blockIdx.x * BN;
  const int wm = wid >> 1, wn = wid & 1;

  const v8f vzero = {0.f, 0.f, 0.f, 0.f, 0.f, 0.f, 0.f, 0.f};
  v8f acc[2][4];
#pragma unroll
  for (int i = 0; i < 2; ++i)
#pragma unroll
    for (int j = 0; j < 4; ++j) acc[i][j] = vzero;

#if HAVE_TDM
  const uint32_t lds_a = (uint32_t)(uintptr_t)(&As[0][0]);
#endif

  for (int kb = 0; kb < K; kb += BK) {
#if HAVE_TDM
    // --- stage A tile [128][32] via Tensor Data Mover (wave 0 only)
    if (wid == 0) {
      tdm_load_tile_2d(lds_a, A + (size_t)m_base * K + kb,
                       (uint32_t)(K - kb), (uint32_t)(M - m_base),
                       (uint64_t)K, (uint32_t)BK, (uint32_t)BM);
    }
#else
    // --- stage A tile [128][32] (16B vector loads, conflict-free padded LDS)
#pragma unroll
    for (int c = 0; c < 2; ++c) {
      int chunk = tid + c * 256;            // 0..511
      int row   = chunk >> 2;
      int k8    = (chunk & 3) << 3;
      *(v8bf*)&As[row][k8] =
          *(const v8bf*)(A + (size_t)(m_base + row) * K + kb + k8);
    }
#endif
    // --- stage B tile [32][128] transposed into Bs[col][k]
#pragma unroll
    for (int c = 0; c < 2; ++c) {
      int chunk = tid + c * 256;            // 0..511
      int k  = chunk >> 4;                  // 0..31
      int n8 = (chunk & 15) << 3;
      v8bf val = *(const v8bf*)(Bm + (size_t)(kb + k) * N + n_base + n8);
#pragma unroll
      for (int i = 0; i < 8; ++i) Bs[n8 + i][k] = val[i];
    }
    if (kb + BK < K) {
      __builtin_prefetch(Bm + (size_t)(kb + BK + (tid >> 4)) * N + n_base, 0, 1);
    }
#if HAVE_TDM
    if (wid == 0) __builtin_amdgcn_s_wait_tensorcnt(0);
#endif
    __syncthreads();

    // --- fragments + 8 WMMAs
    Frag16 af[2], bfr[4];
#pragma unroll
    for (int im = 0; im < 2; ++im) {
      int r = wm * 32 + im * 16 + ln;
      af[im].h[0] = *(const v8bf*)&As[r][o];
      af[im].h[1] = *(const v8bf*)&As[r][o + 16];
    }
#pragma unroll
    for (int jn = 0; jn < 4; ++jn) {
      int cix = wn * 64 + jn * 16 + ln;
      bfr[jn].h[0] = *(const v8bf*)&Bs[cix][o];
      bfr[jn].h[1] = *(const v8bf*)&Bs[cix][o + 16];
    }
#pragma unroll
    for (int im = 0; im < 2; ++im)
#pragma unroll
      for (int jn = 0; jn < 4; ++jn)
        acc[im][jn] = wmma_bf16(af[im], bfr[jn], acc[im][jn]);
    __syncthreads();
  }

  // --- epilogue: bias (+ReLU), store f32 or bf16
#pragma unroll
  for (int jn = 0; jn < 4; ++jn) {
    const int col = n_base + wn * 64 + jn * 16 + ln;
    const float bv = bias[col];
#pragma unroll
    for (int im = 0; im < 2; ++im) {
#pragma unroll
      for (int r = 0; r < 8; ++r) {
        int row = m_base + wm * 32 + im * 16 + r + 8 * hi;
        float v = acc[im][jn][r] + bv;
        if (RELU) v = fmaxf(v, 0.f);
        if (OUTBF) ((bf16*)Cout)[(size_t)row * N + col] = f2bf(v);
        else       ((float*)Cout)[(size_t)row * N + col] = v;
      }
    }
  }
}

// ---------------------------------------------------------------- attention
// One block = one (b, h, 64-query tile); 4 waves x 16 queries.
// Flash-style streaming over keys in chunks of 32 with online softmax.
__global__ void __launch_bounds__(128)
attn_kernel(const bf16* __restrict__ Q, const bf16* __restrict__ Km,
            const bf16* __restrict__ Vm, bf16* __restrict__ Ctx) {
  const int bid = blockIdx.x;
  const int qt = bid & 31;          // S/64 = 32 query tiles
  const int h  = (bid >> 5) & 15;
  const int b  = bid >> 9;
  const int tid  = threadIdx.x;
  const int wid  = tid >> 5;
  const int lane = tid & 31;
  const int ln   = lane & 15;
  const int hi   = lane >> 4;
  const int o    = hi << 3;

  __shared__ __align__(16) bf16  Vs[64][40];       // V chunk transposed [dh][key]
  __shared__ float               Ls[4][16][33];    // per-wave logits
  __shared__ __align__(16) bf16  Ps[4][16][40];    // per-wave P in A-frag layout
  __shared__ float               fac[4][16];       // per-row rescale factors

  const size_t base = ((size_t)b * TS) * TD + (size_t)h * TDH;
  const int q0 = qt * 64 + wid * 16;

  // Q fragments: 16 queries x 64 dh = 2 K-steps (persist whole loop)
  Frag16 qf[2];
  {
    const bf16* qrow = Q + base + (size_t)(q0 + ln) * TD;
#pragma unroll
    for (int kk = 0; kk < 2; ++kk) {
      qf[kk].h[0] = *(const v8bf*)(qrow + kk * 32 + o);
      qf[kk].h[1] = *(const v8bf*)(qrow + kk * 32 + o + 16);
    }
  }

  const v8f vzero = {0.f, 0.f, 0.f, 0.f, 0.f, 0.f, 0.f, 0.f};
  v8f ov[4];
#pragma unroll
  for (int t = 0; t < 4; ++t) ov[t] = vzero;
  float m = -3.4e38f, l = 0.f;  // per-row stats live in lanes 0..15

  for (int kc = 0; kc < TS; kc += 32) {
    __syncthreads();  // protect Vs reuse
    // stage V chunk [32 keys][64 dh] -> Vs[dh][key] (block-shared)
#pragma unroll
    for (int c = 0; c < 2; ++c) {
      int chunk = tid + c * 128;        // 0..255
      int key   = chunk >> 3;           // 0..31
      int d8    = (chunk & 7) << 3;
      v8bf val = *(const v8bf*)(Vm + base + (size_t)(kc + key) * TD + d8);
#pragma unroll
      for (int i = 0; i < 8; ++i) Vs[d8 + i][key] = val[i];
    }
    __syncthreads();

    // logits S = (Q Kt) * 1/sqrt(DH): 2 N-tiles x 2 K-steps of WMMA
#pragma unroll
    for (int j = 0; j < 2; ++j) {
      const bf16* krow = Km + base + (size_t)(kc + j * 16 + ln) * TD;
      Frag16 kf0, kf1;  // B-frag: lane = key column, dh runs {8h, 8h+16}
      kf0.h[0] = *(const v8bf*)(krow + o);
      kf0.h[1] = *(const v8bf*)(krow + o + 16);
      kf1.h[0] = *(const v8bf*)(krow + 32 + o);
      kf1.h[1] = *(const v8bf*)(krow + 32 + o + 16);
      v8f cacc = vzero;
      cacc = wmma_bf16(qf[0], kf0, cacc);
      cacc = wmma_bf16(qf[1], kf1, cacc);
#pragma unroll
      for (int r = 0; r < 8; ++r)
        Ls[wid][r + 8 * hi][j * 16 + ln] = cacc[r] * 0.125f;
    }
    __syncthreads();

    // online softmax row pass: lane r handles row r of this wave's tile
    if (lane < 16) {
      const int r = lane;
      float mc = -3.4e38f;
      for (int i = 0; i < 32; ++i) mc = fmaxf(mc, Ls[wid][r][i]);
      float mn = fmaxf(m, mc);
      float f  = __expf(m - mn);
      float s  = 0.f;
      for (int i = 0; i < 32; ++i) {
        float p = __expf(Ls[wid][r][i] - mn);
        s += p;
        Ps[wid][r][i] = f2bf(p);
      }
      l = l * f + s;
      m = mn;
      fac[wid][r] = f;
    }
    __syncthreads();

    // P(16x32) x V(32x64): rescale accumulators then 4 WMMAs
    Frag16 pf;
    pf.h[0] = *(const v8bf*)&Ps[wid][ln][o];
    pf.h[1] = *(const v8bf*)&Ps[wid][ln][o + 16];
    float fs[8];
#pragma unroll
    for (int r = 0; r < 8; ++r) fs[r] = fac[wid][r + 8 * hi];
#pragma unroll
    for (int t = 0; t < 4; ++t) {
      Frag16 vf;
      vf.h[0] = *(const v8bf*)&Vs[t * 16 + ln][o];
      vf.h[1] = *(const v8bf*)&Vs[t * 16 + ln][o + 16];
#pragma unroll
      for (int r = 0; r < 8; ++r) ov[t][r] *= fs[r];
      ov[t] = wmma_bf16(pf, vf, ov[t]);
    }
  }

  // finalize: divide by l, write ctx back in [B,S,D] layout (bf16)
  if (lane < 16) fac[wid][lane] = 1.f / l;
  __syncthreads();
  float fs[8];
#pragma unroll
  for (int r = 0; r < 8; ++r) fs[r] = fac[wid][r + 8 * hi];
#pragma unroll
  for (int t = 0; t < 4; ++t)
#pragma unroll
    for (int r = 0; r < 8; ++r)
      Ctx[base + (size_t)(q0 + r + 8 * hi) * TD + t * 16 + ln] =
          f2bf(ov[t][r] * fs[r]);
}

// ---------------------------------------------------------------- add + LN
// Out = LN(A + Bd) * g + be ; optional bf16 mirror. Safe when A == Out.
__global__ void __launch_bounds__(256)
add_ln_kernel(const float* A, const float* __restrict__ Bd,
              const float* __restrict__ g, const float* __restrict__ be,
              float* Out, bf16* __restrict__ OutB) {
  const int row = blockIdx.x;
  const int tid = threadIdx.x;
  const float* a  = A + (size_t)row * TD;
  const float* bb = Bd + (size_t)row * TD;
  float xv[4];
  float s = 0.f, s2 = 0.f;
#pragma unroll
  for (int i = 0; i < 4; ++i) {
    int c = tid + i * 256;
    xv[i] = a[c] + bb[c];
    s += xv[i];
    s2 += xv[i] * xv[i];
  }
  __shared__ float r1[256], r2[256];
  r1[tid] = s; r2[tid] = s2;
  __syncthreads();
  for (int off = 128; off > 0; off >>= 1) {
    if (tid < off) { r1[tid] += r1[tid + off]; r2[tid] += r2[tid + off]; }
    __syncthreads();
  }
  const float mean = r1[0] * (1.f / TD);
  const float var  = r2[0] * (1.f / TD) - mean * mean;
  const float rstd = rsqrtf(var + 1e-6f);
  float* outrow = Out + (size_t)row * TD;
  bf16* outbrow = OutB ? (OutB + (size_t)row * TD) : (bf16*)0;
#pragma unroll
  for (int i = 0; i < 4; ++i) {
    int c = tid + i * 256;
    float y = (xv[i] - mean) * rstd * g[c] + be[c];
    outrow[c] = y;
    if (outbrow) outbrow[c] = f2bf(y);
  }
}

// ---------------------------------------------------------------- launch
extern "C" void kernel_launch(void* const* d_in, const int* in_sizes, int n_in,
                              void* d_out, int out_size, void* d_ws,
                              size_t ws_size, hipStream_t stream) {
  (void)in_sizes; (void)n_in; (void)out_size; (void)ws_size;
  const float* x   = (const float*)d_in[0];
  const float* wq  = (const float*)d_in[1];
  const float* bq  = (const float*)d_in[2];
  const float* wk  = (const float*)d_in[3];
  const float* bk  = (const float*)d_in[4];
  const float* wv  = (const float*)d_in[5];
  const float* bv  = (const float*)d_in[6];
  const float* wo  = (const float*)d_in[7];
  const float* bo  = (const float*)d_in[8];
  const float* w1  = (const float*)d_in[9];
  const float* b1  = (const float*)d_in[10];
  const float* w2  = (const float*)d_in[11];
  const float* b2  = (const float*)d_in[12];
  const float* g1  = (const float*)d_in[13];
  const float* be1 = (const float*)d_in[14];
  const float* g2  = (const float*)d_in[15];
  const float* be2 = (const float*)d_in[16];

  constexpr size_t MB = 1ull << 20;
  char* ws = (char*)d_ws;
  bf16* xb   = (bf16*)(ws + 0 * MB);     // 16 MB  [dead after QKV gemms]
  bf16* qb   = (bf16*)(ws + 16 * MB);    // 16 MB  [dead after attention]
  bf16* kb   = (bf16*)(ws + 32 * MB);    // 16 MB
  bf16* vbuf = (bf16*)(ws + 48 * MB);    // 16 MB
  bf16* h1b  = (bf16*)(ws + 0 * MB);     // 64 MB, aliases xb/qb/kb/vbuf
  bf16* wqb  = (bf16*)(ws + 64 * MB);    // 2 MB each
  bf16* wkb  = (bf16*)(ws + 66 * MB);
  bf16* wvb  = (bf16*)(ws + 68 * MB);
  bf16* wob  = (bf16*)(ws + 70 * MB);
  bf16* w1b  = (bf16*)(ws + 72 * MB);    // 8 MB
  bf16* w2b  = (bf16*)(ws + 80 * MB);    // 8 MB
  bf16* ctxb = (bf16*)(ws + 88 * MB);    // 16 MB
  float* resid = (float*)(ws + 104 * MB);  // 32 MB (attn_out, then ffn_out)
  bf16* ln1b = (bf16*)(ws + 136 * MB);   // 16 MB   -> total 152 MB
  float* out = (float*)d_out;

  const long nBSD = (long)TB * TS * TD;        // 8M
  const long nDD  = (long)TD * TD;             // 1M
  const long nDF  = (long)TD * TF;             // 4M

  cvt_f2bf4<<<(unsigned)(nBSD / 1024), 256, 0, stream>>>(x, xb, nBSD);
  cvt_f2bf4<<<(unsigned)(nDD / 1024), 256, 0, stream>>>(wq, wqb, nDD);
  cvt_f2bf4<<<(unsigned)(nDD / 1024), 256, 0, stream>>>(wk, wkb, nDD);
  cvt_f2bf4<<<(unsigned)(nDD / 1024), 256, 0, stream>>>(wv, wvb, nDD);
  cvt_f2bf4<<<(unsigned)(nDD / 1024), 256, 0, stream>>>(wo, wob, nDD);
  cvt_f2bf4<<<(unsigned)(nDF / 1024), 256, 0, stream>>>(w1, w1b, nDF);
  cvt_f2bf4<<<(unsigned)(nDF / 1024), 256, 0, stream>>>(w2, w2b, nDF);

  const int M = TB * TS;                        // 8192
  dim3 gD(TD / 128, M / 128);                   // (8, 64)
  dim3 gF(TF / 128, M / 128);                   // (32, 64)

  // QKV projections (bf16 out for attention)
  gemm_bf16k<false, true><<<gD, 256, 0, stream>>>(xb, wqb, bq, qb, M, TD, TD);
  gemm_bf16k<false, true><<<gD, 256, 0, stream>>>(xb, wkb, bk, kb, M, TD, TD);
  gemm_bf16k<false, true><<<gD, 256, 0, stream>>>(xb, wvb, bv, vbuf, M, TD, TD);

  // attention -> ctx (bf16, [B,S,D])
  attn_kernel<<<TB * THD * (TS / 64), 128, 0, stream>>>(qb, kb, vbuf, ctxb);

  // out projection (f32) + residual LN1 (f32 out1 in d_out, bf16 mirror)
  gemm_bf16k<false, false><<<gD, 256, 0, stream>>>(ctxb, wob, bo, resid, M, TD, TD);
  add_ln_kernel<<<M, 256, 0, stream>>>(x, resid, g1, be1, out, ln1b);

  // FFN: relu(out1 @ w1 + b1) @ w2 + b2
  gemm_bf16k<true, true><<<gF, 256, 0, stream>>>(ln1b, w1b, b1, h1b, M, TF, TD);
  gemm_bf16k<false, false><<<gD, 256, 0, stream>>>(h1b, w2b, b2, resid, M, TD, TF);

  // residual LN2 in-place on d_out
  add_ln_kernel<<<M, 256, 0, stream>>>(out, resid, g2, be2, out, (bf16*)0);
}
